// Build_Simulator_44092134261028
// MI455X (gfx1250) — compile-verified, matrix operand order checked by reading, not run on hardware
//
#include <hip/hip_runtime.h>
#include <hip/hip_bf16.h>
#include <stdint.h>

// ---------------------------------------------------------------------------
// Problem constants (from reference): C=10 celltypes, N=1000 cells, G=18000
// genes, 500 total sampled rows, Dense is [4,64]@[64,10].
// ---------------------------------------------------------------------------
#define CTYPES      10
#define NCELLS      1000
#define GMAX        18000
#define TOTAL_DRAWS 500
#define LN_EPS      1e-3f
#define ALPHA_EPS   1e-6f

#define AS1 __attribute__((address_space(1)))
#define AS3 __attribute__((address_space(3)))

typedef __attribute__((ext_vector_type(2)))  float    v2f;
typedef __attribute__((ext_vector_type(8)))  float    v8f;
typedef __attribute__((ext_vector_type(16))) _Float16 v16h;
typedef __attribute__((ext_vector_type(4)))  int      v4i;

#if defined(__has_builtin)
#if __has_builtin(__builtin_amdgcn_wmma_f32_16x16x4_f32)
#define HAVE_WMMA_F32X4 1
#endif
#if __has_builtin(__builtin_amdgcn_global_load_async_to_lds_b128) && \
    __has_builtin(__builtin_amdgcn_s_wait_asynccnt)
#define HAVE_ASYNC_LDS 1
#endif
#endif

// ------------------------------ threefry2x32 --------------------------------
__device__ __forceinline__ uint32_t rotl32(uint32_t v, int n) {
    return (v << n) | (v >> (32 - n));
}

__device__ void tf2x32(uint32_t k0, uint32_t k1, uint32_t c0, uint32_t c1,
                       uint32_t& o0, uint32_t& o1) {
    const uint32_t ks0 = k0, ks1 = k1, ks2 = 0x1BD11BDAu ^ k0 ^ k1;
    uint32_t x0 = c0 + ks0, x1 = c1 + ks1;
    const int RA[4] = {13, 15, 26, 6};
    const int RB[4] = {17, 29, 16, 24};
#pragma unroll
    for (int j = 0; j < 4; ++j) { x0 += x1; x1 = rotl32(x1, RA[j]); x1 ^= x0; }
    x0 += ks1; x1 += ks2 + 1u;
#pragma unroll
    for (int j = 0; j < 4; ++j) { x0 += x1; x1 = rotl32(x1, RB[j]); x1 ^= x0; }
    x0 += ks2; x1 += ks0 + 2u;
#pragma unroll
    for (int j = 0; j < 4; ++j) { x0 += x1; x1 = rotl32(x1, RA[j]); x1 ^= x0; }
    x0 += ks0; x1 += ks1 + 3u;
#pragma unroll
    for (int j = 0; j < 4; ++j) { x0 += x1; x1 = rotl32(x1, RB[j]); x1 ^= x0; }
    x0 += ks1; x1 += ks2 + 4u;
#pragma unroll
    for (int j = 0; j < 4; ++j) { x0 += x1; x1 = rotl32(x1, RA[j]); x1 ^= x0; }
    x0 += ks2; x1 += ks0 + 5u;
    o0 = x0; o1 = x1;
}

__device__ __forceinline__ uint32_t rbits(uint32_t k0, uint32_t k1, uint32_t& ctr) {
    uint32_t o0, o1;
    tf2x32(k0, k1, ctr++, 0u, o0, o1);
    return o0;
}

// uniform in (0,1): never exactly 0 or 1, safe for logs
__device__ __forceinline__ float u01(uint32_t b) {
    return ((float)(b >> 8) + 0.5f) * 5.9604645e-8f;  // 2^-24
}

__device__ float normal_sample(uint32_t k0, uint32_t k1, uint32_t& ctr) {
    float u1 = u01(rbits(k0, k1, ctr));
    float u2 = u01(rbits(k0, k1, ctr));
    return sqrtf(-2.0f * logf(u1)) * cosf(6.2831853071795865f * u2);
}

// Marsaglia-Tsang with a<1 boost (structure matches jax.random.gamma)
__device__ float gamma_sample(uint32_t k0, uint32_t k1, uint32_t& ctr, float a) {
    float boost = 1.0f;
    if (a < 1.0f) {
        boost = powf(u01(rbits(k0, k1, ctr)), 1.0f / a);
        a += 1.0f;
    }
    const float d = a - (1.0f / 3.0f);
    const float c = rsqrtf(9.0f * d);
    for (int it = 0; it < 256; ++it) {
        float n = normal_sample(k0, k1, ctr);
        float v = 1.0f + c * n;
        if (v <= 0.0f) continue;
        v = v * v * v;
        float u = u01(rbits(k0, k1, ctr));
        if (logf(u) < 0.5f * n * n + d - d * v + d * logf(v)) return boost * d * v;
    }
    return boost * d;
}

// ---------------------------------------------------------------------------
// Kernel A: one wave (32 threads).
//   1) stage x/W in LDS (coalesced, unconditional), alpha = relu(x@W+b)+eps
//      via 16 accumulating V_WMMA_F32_16X16X4_F32 (EXEC all-ones: one wave).
//   2) Dirichlet(alpha0) -> logits -> 500 Gumbel-argmax draws -> counts[10]
//   3) per-celltype Fisher-Yates over 1000 ids in LDS, take first count[c]
//   4) compact selected global row ids (c*1000+n) into workspace
// ---------------------------------------------------------------------------
__global__ __launch_bounds__(32)
void sim_select_kernel(const float* __restrict__ x, const float* __restrict__ W,
                       const float* __restrict__ b,
                       int* __restrict__ hdr, int* __restrict__ rowsOut) {
    __shared__ float sx[4 * 64];    // x staged
    __shared__ float sw[64 * 10];   // W staged
    __shared__ float salpha[CTYPES];
    __shared__ float slogits[CTYPES];
    __shared__ int   scnt[CTYPES];
    __shared__ int   sperm[CTYPES * NCELLS];

    const int lane = threadIdx.x;
    const int lo   = lane & 15;
    const int half = lane >> 4;

    // stage operands into LDS (coalesced, no divergence)
    for (int i = lane; i < 4 * 64; i += 32)  sx[i] = x[i];
    for (int i = lane; i < 64 * 10; i += 32) sw[i] = W[i];
    if (lane < CTYPES) scnt[lane] = 0;
    __syncthreads();

    // clamped indices -> unconditional LDS reads + cndmask (no EXEC branching)
    const int mrow = (lo < 4)  ? lo : 0;   // A row (masked below)
    const int ncol = (lo < 10) ? lo : 0;   // B col (masked below)
    const float am = (lo < 4)  ? 1.0f : 0.0f;
    const float bm = (lo < 10) ? 1.0f : 0.0f;

    v8f acc = {};
#if HAVE_WMMA_F32X4
#pragma unroll
    for (int kk = 0; kk < 16; ++kk) {
        const int k0 = kk * 4 + half * 2;
        v2f av, bv;
        av[0] = am * sx[mrow * 64 + k0];
        av[1] = am * sx[mrow * 64 + k0 + 1];
        bv[0] = bm * sw[(k0)     * 10 + ncol];
        bv[1] = bm * sw[(k0 + 1) * 10 + ncol];
        acc = __builtin_amdgcn_wmma_f32_16x16x4_f32(
            false, av, false, bv, (short)0, acc, false, false);
    }
#else
#pragma unroll
    for (int chunk = 0; chunk < 2; ++chunk) {
        v16h av, bv;
#pragma unroll
        for (int j = 0; j < 16; ++j) {
            const int r = j >> 1;
            const int k = 32 * chunk +
                          ((r < 4) ? 2 * r : 16 + 2 * (r - 4)) +
                          8 * half + (j & 1);
            av[j] = (_Float16)(am * sx[mrow * 64 + k]);
            bv[j] = (_Float16)(bm * sw[k * 10 + ncol]);
        }
        acc = __builtin_amdgcn_wmma_f32_16x16x32_f16(
            false, av, false, bv, (short)0, acc, false, false);
    }
#endif
    // D[m=0][n=lane] lives in acc[0] of lanes 0..15
    if (lane < CTYPES) salpha[lane] = fmaxf(acc[0] + b[lane], 0.0f) + ALPHA_EPS;
    __syncthreads();

    // ---- RNG key derivation (threefry, seed 42) ----------------------------
    const uint32_t key0 = 0u, key1 = 42u;
    uint32_t kd0, kd1, ksub0, ksub1, kg0, kg1, kc0, kc1;
    tf2x32(key0, key1, 0u, 1u, kd0, kd1);     // k_dir
    tf2x32(key0, key1, 2u, 3u, ksub0, ksub1); // k_sub
    tf2x32(kd0, kd1, 0u, 1u, kg0, kg1);       // gamma key
    tf2x32(kd0, kd1, 2u, 3u, kc0, kc1);       // categorical key

    if (lane == 0) {
        float g[CTYPES], s = 0.0f;
        uint32_t ctr = 0;
        for (int i = 0; i < CTYPES; ++i) { g[i] = gamma_sample(kg0, kg1, ctr, salpha[i]); s += g[i]; }
        for (int i = 0; i < CTYPES; ++i) slogits[i] = logf(g[i] / s);
    }
    __syncthreads();

    // ---- 500 categorical draws, lanes parallel over t ----------------------
    for (int t = lane; t < TOTAL_DRAWS; t += 32) {
        int best = 0;
        float bestv = -3.4e38f;
        for (int i = 0; i < CTYPES; ++i) {
            uint32_t o0, o1;
            tf2x32(kc0, kc1, (uint32_t)(t * CTYPES + i), 0u, o0, o1);
            float gum = -logf(-logf(u01(o0)));
            float v = slogits[i] + gum;
            if (v > bestv) { bestv = v; best = i; }
        }
        atomicAdd(&scnt[best], 1);
    }

    // ---- init permutation ids cooperatively --------------------------------
    for (int i = lane; i < CTYPES * NCELLS; i += 32) sperm[i] = i % NCELLS;
    __syncthreads();

    // ---- Fisher-Yates per celltype (10 lanes, independent keys) ------------
    if (lane < CTYPES) {
        uint32_t c0, c1;
        tf2x32(ksub0, ksub1, (uint32_t)(2 * lane), (uint32_t)(2 * lane + 1), c0, c1);
        uint32_t ctr = 0;
        int* p = &sperm[lane * NCELLS];
        for (int i = NCELLS - 1; i >= 1; --i) {
            int j = (int)(rbits(c0, c1, ctr) % (uint32_t)(i + 1));
            int tmp = p[i]; p[i] = p[j]; p[j] = tmp;
        }
    }
    __syncthreads();

    // ---- compact selected rows ---------------------------------------------
    if (lane == 0) {
        int off = 0;
        for (int c = 0; c < CTYPES; ++c) {
            int k = scnt[c];
            if (k > NCELLS) k = NCELLS;
            for (int j = 0; j < k && off < 512; ++j)
                rowsOut[off++] = c * NCELLS + sperm[c * NCELLS + j];
        }
        hdr[0] = off;
    }
}

// ---------------------------------------------------------------------------
// Kernel B: split-K gather-reduce over selected rows. grid = (colTiles, S).
// Each block owns a 1024-column tile and row-group y; writes its partial
// exactly once (deterministic, no atomics).
// CDNA5 path: each wave double-buffers its 512B row-segment through LDS with
// global_load_async_to_lds_b128 + s_wait_asynccnt, hiding the dependent row
// fetch latency. Waves are self-contained (private LDS slots, no barrier in
// the steady-state loop). Fallback: plain float4 loads + prefetch.
// ---------------------------------------------------------------------------
__global__ __launch_bounds__(256)
void rowsum_kernel(const float* __restrict__ scdata, const int* __restrict__ rows,
                   const int* __restrict__ hdr, float* __restrict__ partial,
                   int S, int G) {
    __shared__ int srows[512];
    const int nsel = hdr[0];
    for (int i = threadIdx.x; i < 512; i += 256) srows[i] = (i < nsel) ? rows[i] : 0;
    __syncthreads();

    const int y = blockIdx.y;
    float4 acc = make_float4(0.f, 0.f, 0.f, 0.f);

#if HAVE_ASYNC_LDS
    __shared__ float stage[2][8][128];  // [buf][wave][128 floats] = 8 KB
    const int wave = threadIdx.x >> 5;
    const int lane = threadIdx.x & 31;
    const int g = blockIdx.x * 1024 + wave * 128 + lane * 4;
    if (g >= G) return;  // G%4==0 so g<G implies g+3<G

    float* lds0 = &stage[0][wave][lane * 4];
    float* lds1 = &stage[1][wave][lane * 4];

    int r = y;
    if (r < nsel) {
        __builtin_amdgcn_global_load_async_to_lds_b128(
            (AS1 v4i*)(scdata + (size_t)srows[r] * (size_t)G + g),
            (AS3 v4i*)lds0, 0, 0);
    }
    int cur = 0;
    while (r < nsel) {
        const int rn = r + S;
        if (rn < nsel) {
            __builtin_amdgcn_global_load_async_to_lds_b128(
                (AS1 v4i*)(scdata + (size_t)srows[rn] * (size_t)G + g),
                (AS3 v4i*)(cur ? lds0 : lds1), 0, 0);
            __builtin_amdgcn_s_wait_asynccnt(1);  // current buffer ready
        } else {
            __builtin_amdgcn_s_wait_asynccnt(0);
        }
        const float4 v = *reinterpret_cast<const float4*>(cur ? lds1 : lds0);
        acc.x += v.x; acc.y += v.y; acc.z += v.z; acc.w += v.w;
        cur ^= 1;
        r = rn;
    }
#else
    const int g = blockIdx.x * 1024 + threadIdx.x * 4;
    if (g >= G) return;
    for (int r = y; r < nsel; r += S) {
        const float* row = scdata + (size_t)srows[r] * (size_t)G;
        if (r + S < nsel)
            __builtin_prefetch(scdata + (size_t)srows[r + S] * (size_t)G + g, 0, 0);
        const float4 v = *reinterpret_cast<const float4*>(row + g);
        acc.x += v.x; acc.y += v.y; acc.z += v.z; acc.w += v.w;
    }
#endif
    *reinterpret_cast<float4*>(partial + (size_t)y * (size_t)G + g) = acc;
}

// ---------------------------------------------------------------------------
// Kernel C: reduce S partials, log1p, LayerNorm, min-max. Whole 72KB vector
// is staged in LDS (WGP has 320KB). One block, 1024 threads.
// ---------------------------------------------------------------------------
__device__ float blk_red(float v, volatile float* red, int mode) {
    const int t = threadIdx.x;
    red[t] = v;
    __syncthreads();
    for (int s = 512; s > 0; s >>= 1) {
        if (t < s) {
            float a = red[t], c = red[t + s];
            red[t] = (mode == 0) ? (a + c) : (mode == 1) ? fminf(a, c) : fmaxf(a, c);
        }
        __syncthreads();
    }
    float r = red[0];
    __syncthreads();
    return r;
}

__global__ __launch_bounds__(1024)
void finalize_kernel(const float* __restrict__ partial,
                     const float* __restrict__ gamma, const float* __restrict__ beta,
                     float* __restrict__ out, int S, int G) {
    __shared__ float zbuf[GMAX];   // 72 KB — resident in CDNA5's 320KB WGP LDS
    __shared__ float red[1024];
    const int t = threadIdx.x;

    // reduce split-K partials deterministically + log1p
    float lsum = 0.0f;
    for (int g = t; g < G; g += 1024) {
        float s = 0.0f;
        for (int y = 0; y < S; ++y) s += partial[(size_t)y * (size_t)G + g];
        float z = log1pf(s);
        zbuf[g] = z;
        lsum += z;
    }
    __syncthreads();
    const float mean = blk_red(lsum, red, 0) / (float)G;

    float lvar = 0.0f;
    for (int g = t; g < G; g += 1024) {
        float d = zbuf[g] - mean;
        lvar += d * d;
    }
    const float var = blk_red(lvar, red, 0) / (float)G;
    const float rinv = rsqrtf(var + LN_EPS);

    float lmin = 3.4e38f, lmax = -3.4e38f;
    for (int g = t; g < G; g += 1024) {
        float yv = (zbuf[g] - mean) * rinv * gamma[g] + beta[g];
        zbuf[g] = yv;
        lmin = fminf(lmin, yv);
        lmax = fmaxf(lmax, yv);
    }
    __syncthreads();
    const float lov = blk_red(lmin, red, 1);
    const float hiv = blk_red(lmax, red, 2);
    const float denom = hiv - lov;

    for (int g = t; g < G; g += 1024)
        out[g] = (denom == 0.0f) ? 0.0f : (zbuf[g] - lov) / denom;
}

// ---------------------------------------------------------------------------
extern "C" void kernel_launch(void* const* d_in, const int* in_sizes, int n_in,
                              void* d_out, int out_size, void* d_ws, size_t ws_size,
                              hipStream_t stream) {
    const float* x      = (const float*)d_in[0];
    const float* W      = (const float*)d_in[1];
    const float* b      = (const float*)d_in[2];
    const float* scdata = (const float*)d_in[3];
    const float* gamma  = (const float*)d_in[4];
    const float* beta   = (const float*)d_in[5];
    float* out = (float*)d_out;

    const int G = in_sizes[4];  // 18000

    uint8_t* ws = (uint8_t*)d_ws;
    int*   hdr     = (int*)ws;            // [0]: nsel
    int*   rows    = (int*)(ws + 64);     // up to 512 row ids
    const size_t partialOff = 4096;
    float* partial = (float*)(ws + partialOff);

    // split-K factor limited by workspace
    int S = 1;
    if (ws_size > partialOff + (size_t)G * 4) {
        S = (int)((ws_size - partialOff) / ((size_t)G * 4));
        if (S > 8) S = 8;
        if (S < 1) S = 1;
    }

    sim_select_kernel<<<1, 32, 0, stream>>>(x, W, b, hdr, rows);

    dim3 gridB((G + 1023) / 1024, S);
    rowsum_kernel<<<gridB, 256, 0, stream>>>(scdata, rows, hdr, partial, S, G);

    finalize_kernel<<<1, 1024, 0, stream>>>(partial, gamma, beta, out, S, G);
}
